// ODEFunc_52699248722474
// MI455X (gfx1250) — matrix operand order repaired and checked by reading, not compiled, not run on hardware
//
#include <hip/hip_runtime.h>
#include <hip/hip_bf16.h>
#include <math.h>

// ---------------- problem constants ----------------
constexpr int CB   = 2;            // batch
constexpr int CT   = 2048;         // seq len
constexpr int CD   = 1024;         // model dim
constexpr int CNH  = 16;           // heads
constexpr int CDH  = 64;           // head dim
constexpr int CHID = 4096;         // mlp hidden
constexpr int CM   = CB * CT;      // total rows = 4096

typedef __attribute__((ext_vector_type(16))) __bf16 v16bf;
typedef __attribute__((ext_vector_type(8)))  __bf16 v8bf;
typedef __attribute__((ext_vector_type(8)))  float  v8f;
typedef __attribute__((ext_vector_type(4)))  unsigned int v4u;
typedef __attribute__((ext_vector_type(8)))  int v8i;
typedef __attribute__((ext_vector_type(4)))  int v4i;

// ---- CDNA5 WMMA fragment loaders (wave32) ----
// A-matrix 16x32 bf16: lane-half h holds K in {8h..8h+7} U {16+8h..16+8h+7}
__device__ __forceinline__ v16bf load_afrag(const __bf16* rowp, int half) {
    union { v16bf v; v8bf h[2]; } f;
    f.h[0] = *reinterpret_cast<const v8bf*>(rowp + 8 * half);
    f.h[1] = *reinterpret_cast<const v8bf*>(rowp + 16 + 8 * half);
    return f.v;
}
// B-matrix 32x16 bf16: lane-half h holds K in {8h..8h+7} U {16+8h..16+8h+7}
__device__ __forceinline__ v16bf load_bfrag(const __bf16* rowp, int half) {
    union { v16bf v; v8bf h[2]; } f;
    f.h[0] = *reinterpret_cast<const v8bf*>(rowp + 8 * half);
    f.h[1] = *reinterpret_cast<const v8bf*>(rowp + 16 + 8 * half);
    return f.v;
}

// ---- Tensor Data Mover: DMA a 2D bf16 tile (tile_k x tile_rows) into LDS ----
// D# layout per CDNA5 ISA ch.8 (group0: count/lds/global/type; group1: dims).
__device__ __forceinline__ void tdm_load_2d_bf16(
    unsigned lds_off, const void* gtile,
    unsigned tile_k, unsigned tile_rows,
    unsigned tensor_k, unsigned tensor_rows, unsigned row_stride_elems) {
    unsigned long long ga = (unsigned long long)gtile;
    v4u g0;
    g0[0] = 1u;                                           // count=1
    g0[1] = lds_off;                                      // LDS byte address
    g0[2] = (unsigned)(ga & 0xFFFFFFFFu);                 // global addr lo
    g0[3] = (unsigned)((ga >> 32) & 0x01FFFFFFu) | (2u << 30);  // addr hi | type=2
    v8i g1;
    g1[0] = (int)(1u << 16);                              // data_size=1 (2B), wg_mask=0
    g1[1] = (int)((tensor_k & 0xFFFFu) << 16);            // tensor_dim0[15:0]
    g1[2] = (int)(((tensor_k >> 16) & 0xFFFFu) |
                  ((tensor_rows & 0xFFFFu) << 16));       // dim0 hi | dim1 lo
    g1[3] = (int)(((tensor_rows >> 16) & 0xFFFFu) |
                  ((tile_k & 0xFFFFu) << 16));            // dim1 hi | tile_dim0
    g1[4] = (int)(tile_rows & 0xFFFFu);                   // tile_dim1 (tile_dim2=0)
    g1[5] = (int)row_stride_elems;                        // tensor_dim0_stride lo32
    g1[6] = 0;                                            // stride hi | dim1_stride lo
    g1[7] = 0;
    v4i z4 = {0, 0, 0, 0};
#if __clang_major__ >= 23
    v8i z8 = {0, 0, 0, 0, 0, 0, 0, 0};
    __builtin_amdgcn_tensor_load_to_lds(g0, g1, z4, z4, z8, 0);
#else
    __builtin_amdgcn_tensor_load_to_lds(g0, g1, z4, z4, 0);
#endif
}

// ---------------- fp32 -> bf16 cast ----------------
__global__ void cast_bf16_kernel(const float* __restrict__ in,
                                 __bf16* __restrict__ out, long n) {
    long i = (long)blockIdx.x * blockDim.x + threadIdx.x;
    long stride = (long)gridDim.x * blockDim.x;
    for (; i < n; i += stride) out[i] = (__bf16)in[i];
}

// ---------------- RMSNorm (optionally x = X + A) ----------------
__global__ __launch_bounds__(256) void rmsnorm_kernel(
    const float* __restrict__ X, const float* __restrict__ Aadd,
    const float* __restrict__ scale, __bf16* __restrict__ out, int Dn) {
    int row = blockIdx.x;
    const float* x = X + (size_t)row * Dn;
    const float* a = Aadd ? Aadd + (size_t)row * Dn : nullptr;
    __shared__ float red[8];
    __shared__ float rinv_s;
    float ss = 0.f;
    for (int i = threadIdx.x; i < Dn; i += blockDim.x) {
        float v = x[i] + (a ? a[i] : 0.f);
        ss += v * v;
    }
    for (int off = 16; off; off >>= 1) ss += __shfl_xor(ss, off, 32);
    if ((threadIdx.x & 31) == 0) red[threadIdx.x >> 5] = ss;
    __syncthreads();
    if (threadIdx.x == 0) {
        float t = 0.f;
        for (int i = 0; i < 8; ++i) t += red[i];
        rinv_s = rsqrtf(t / (float)Dn + 1e-6f);
    }
    __syncthreads();
    float rinv = rinv_s;
    for (int i = threadIdx.x; i < Dn; i += blockDim.x) {
        float v = x[i] + (a ? a[i] : 0.f);
        out[(size_t)row * Dn + i] = (__bf16)(scale[i] * v * rinv);
    }
}

// ------- tiled WMMA GEMM with TDM double-buffered staging -------
// C[M,N] = A[M,K] * W[N,K]^T ; tiles: 128x128xBK64, 8 waves, wave tile 64x32
// EPI: 0 = store bf16; 1 = store fp32; 2 = +bias, exact GELU, store bf16;
//      3 = +bias + extra[M,N], store fp32
template <int EPI>
__global__ __launch_bounds__(256) void gemm_bf16_wmma(
    const __bf16* __restrict__ Amat, const __bf16* __restrict__ Wmat,
    const float* __restrict__ bias, const float* __restrict__ extra,
    void* __restrict__ outp, int M, int N, int K) {
    __shared__ alignas(128) __bf16 As[2][128][64];   // 32 KB
    __shared__ alignas(128) __bf16 Bs[2][128][64];   // 32 KB

    const int tid  = threadIdx.x;
    const int wave = tid >> 5;
    const int lane = tid & 31;
    const int waveM = wave & 1;   // 2 waves along M
    const int waveN = wave >> 1;  // 4 waves along N
    const int half = lane >> 4;
    const int lm   = lane & 15;
    const int bm = blockIdx.x * 128;
    const int bn = blockIdx.y * 128;

    const unsigned ldsA0 = (unsigned)(unsigned long long)(void*)&As[0][0][0];
    const unsigned ldsA1 = (unsigned)(unsigned long long)(void*)&As[1][0][0];
    const unsigned ldsB0 = (unsigned)(unsigned long long)(void*)&Bs[0][0][0];
    const unsigned ldsB1 = (unsigned)(unsigned long long)(void*)&Bs[1][0][0];

    const int nkt = K / 64;
    v8f acc[4][2] = {};

    // preload tile 0 via TDM (one wave issues the DMA for the workgroup)
    if (wave == 0) {
        tdm_load_2d_bf16(ldsA0, Amat + (size_t)bm * K, 64, 128,
                         (unsigned)K, (unsigned)M, (unsigned)K);
        tdm_load_2d_bf16(ldsB0, Wmat + (size_t)bn * K, 64, 128,
                         (unsigned)K, (unsigned)N, (unsigned)K);
    }

    for (int kt = 0; kt < nkt; ++kt) {
        const int cur = kt & 1;
        if (wave == 0) {
            if (kt + 1 < nkt) {
                const int k0n = (kt + 1) * 64;
                tdm_load_2d_bf16(cur ? ldsA0 : ldsA1,
                                 Amat + (size_t)bm * K + k0n, 64, 128,
                                 (unsigned)K, (unsigned)M, (unsigned)K);
                tdm_load_2d_bf16(cur ? ldsB0 : ldsB1,
                                 Wmat + (size_t)bn * K + k0n, 64, 128,
                                 (unsigned)K, (unsigned)N, (unsigned)K);
                __builtin_amdgcn_s_wait_tensorcnt(2);  // current tile landed
            } else {
                __builtin_amdgcn_s_wait_tensorcnt(0);
            }
        }
        __syncthreads();   // current buffer visible to all waves

        for (int s = 0; s < 2; ++s) {
            v16bf af[4], bfv[2];
            for (int mt = 0; mt < 4; ++mt)
                af[mt] = load_afrag(&As[cur][waveM * 64 + mt * 16 + lm][s * 32], half);
            for (int nt = 0; nt < 2; ++nt)
                bfv[nt] = load_bfrag(&Bs[cur][waveN * 32 + nt * 16 + lm][s * 32], half);
            for (int mt = 0; mt < 4; ++mt)
                for (int nt = 0; nt < 2; ++nt)
                    acc[mt][nt] = __builtin_amdgcn_wmma_f32_16x16x32_bf16(
                        false, af[mt], false, bfv[nt], (short)0, acc[mt][nt],
                        false, false);
        }
        __syncthreads();   // all reads done before DMA overwrites this buffer
    }

    // epilogue (C/D layout: m = r + 8*half, n = lm)
    for (int mt = 0; mt < 4; ++mt) {
        for (int nt = 0; nt < 2; ++nt) {
            int col = bn + waveN * 32 + nt * 16 + lm;
            float bv = (EPI == 2 || EPI == 3) ? bias[col] : 0.f;
            for (int r = 0; r < 8; ++r) {
                int row = bm + waveM * 64 + mt * 16 + r + half * 8;
                size_t idx = (size_t)row * N + col;
                float v = acc[mt][nt][r];
                if constexpr (EPI == 0) {
                    ((__bf16*)outp)[idx] = (__bf16)v;
                } else if constexpr (EPI == 1) {
                    ((float*)outp)[idx] = v;
                } else if constexpr (EPI == 2) {
                    v += bv;
                    v = 0.5f * v * (1.f + erff(v * 0.70710678118f));
                    ((__bf16*)outp)[idx] = (__bf16)v;
                } else {
                    ((float*)outp)[idx] = v + bv + extra[idx];
                }
            }
        }
    }
}

// ---------------- flash attention (bf16 WMMA, online softmax) --------------
// grid: (CT/64, CB*CNH); block: 128 (4 waves, 16 queries each)
__global__ __launch_bounds__(128) void flash_attn_kernel(
    const __bf16* __restrict__ Q, const __bf16* __restrict__ Kt,
    const __bf16* __restrict__ V, const unsigned char* __restrict__ mask,
    __bf16* __restrict__ Y) {
    __shared__ alignas(32) __bf16 Ks[64][64];       // [key][d]
    __shared__ alignas(32) __bf16 Vt[64][64];       // [d][key] (transposed)
    __shared__ alignas(32) __bf16 Ps[4][16][64];    // per-wave P tile

    const int tid  = threadIdx.x;
    const int w    = tid >> 5;
    const int lane = tid & 31;
    const int half = lane >> 4;
    const int lm   = lane & 15;

    const int qtile = blockIdx.x;
    const int b = blockIdx.y / CNH;
    const int h = blockIdx.y % CNH;
    const size_t rowbase = (size_t)b * CT;
    const int hcol = h * CDH;

    // Q fragments for this wave's 16 query rows (A layout), kept in regs
    const int qrowA = qtile * 64 + w * 16 + lm;
    v16bf qf[2];
    for (int s = 0; s < 2; ++s)
        qf[s] = load_afrag(Q + (rowbase + qrowA) * CD + hcol + s * 32, half);

    v8f yacc[4] = {};
    float runmax[8], runsum[8];
    for (int r = 0; r < 8; ++r) { runmax[r] = -1e30f; runsum[r] = 0.f; }

    const float scale = 0.125f;  // 1/sqrt(64)

    for (int kt = 0; kt < CT / 64; ++kt) {
        // ---- stage K tile and transposed V tile ----
        {
            int krow = tid >> 1;
            int ch = (tid & 1);
            const __bf16* krp = Kt + (rowbase + kt * 64 + krow) * CD + hcol + ch * 32;
            *reinterpret_cast<v16bf*>(&Ks[krow][ch * 32]) =
                *reinterpret_cast<const v16bf*>(krp);
            *reinterpret_cast<v16bf*>(&Ks[krow][ch * 32 + 16]) =
                *reinterpret_cast<const v16bf*>(krp + 16);
            const __bf16* vrp = V + (rowbase + kt * 64 + krow) * CD + hcol + ch * 32;
            union { v16bf v; __bf16 e[16]; } t0, t1;
            t0.v = *reinterpret_cast<const v16bf*>(vrp);
            t1.v = *reinterpret_cast<const v16bf*>(vrp + 16);
            for (int i = 0; i < 16; ++i) {
                Vt[ch * 32 + i][krow]      = t0.e[i];
                Vt[ch * 32 + 16 + i][krow] = t1.e[i];
            }
        }
        __syncthreads();

        // ---- S = Q * K^T for 64 keys (4 N-tiles x 2 K-steps) ----
        float smat[4][8];
        for (int nt = 0; nt < 4; ++nt) {
            v8f sacc = {};
            for (int s = 0; s < 2; ++s) {
                v16bf bf = load_bfrag(&Ks[nt * 16 + lm][s * 32], half);
                sacc = __builtin_amdgcn_wmma_f32_16x16x32_bf16(
                    false, qf[s], false, bf, (short)0, sacc, false, false);
            }
            int key = kt * 64 + nt * 16 + lm;
            bool mk = mask[(size_t)b * CT + key] != 0;
            for (int r = 0; r < 8; ++r)
                smat[nt][r] = mk ? sacc[r] * scale : -1e30f;
        }

        // ---- online softmax (row reductions across 16-lane half) ----
        float tmax[8];
        for (int r = 0; r < 8; ++r) {
            float m = smat[0][r];
            for (int nt = 1; nt < 4; ++nt) m = fmaxf(m, smat[nt][r]);
            for (int off = 1; off < 16; off <<= 1)
                m = fmaxf(m, __shfl_xor(m, off, 32));
            tmax[r] = m;
        }
        for (int r = 0; r < 8; ++r) {
            float nm = fmaxf(runmax[r], tmax[r]);
            float alpha = expf(runmax[r] - nm);
            runsum[r] *= alpha;
            for (int dt = 0; dt < 4; ++dt) yacc[dt][r] *= alpha;
            runmax[r] = nm;
        }
        float psum[8];
        for (int r = 0; r < 8; ++r) psum[r] = 0.f;
        for (int nt = 0; nt < 4; ++nt) {
            for (int r = 0; r < 8; ++r) {
                float p = expf(smat[nt][r] - runmax[r]);
                psum[r] += p;
                Ps[w][r + 8 * half][nt * 16 + lm] = (__bf16)p;
            }
        }
        for (int r = 0; r < 8; ++r) {
            float s = psum[r];
            for (int off = 1; off < 16; off <<= 1) s += __shfl_xor(s, off, 32);
            runsum[r] += s;
        }

        // ---- Y += P * V (A frags from LDS, B frags from transposed V) ----
        for (int s = 0; s < 2; ++s) {
            v16bf pf = load_afrag(&Ps[w][lm][s * 32], half);
            for (int dt = 0; dt < 4; ++dt) {
                v16bf vf = load_bfrag(&Vt[dt * 16 + lm][s * 32], half);
                yacc[dt] = __builtin_amdgcn_wmma_f32_16x16x32_bf16(
                    false, pf, false, vf, (short)0, yacc[dt], false, false);
            }
        }
        __syncthreads();
    }

    // ---- normalize and store (C/D layout) ----
    for (int dt = 0; dt < 4; ++dt) {
        for (int r = 0; r < 8; ++r) {
            int q = qtile * 64 + w * 16 + r + 8 * half;
            int col = hcol + dt * 16 + lm;
            float denom = runsum[r];
            float v = denom > 0.f ? yacc[dt][r] / denom : 0.f;
            Y[(rowbase + q) * CD + col] = (__bf16)v;
        }
    }
}

// ---------------- host-side orchestration ----------------
extern "C" void kernel_launch(void* const* d_in, const int* in_sizes, int n_in,
                              void* d_out, int out_size, void* d_ws, size_t ws_size,
                              hipStream_t stream) {
    (void)in_sizes; (void)n_in; (void)out_size; (void)ws_size;
    const float* Hh   = (const float*)d_in[1];
    const unsigned char* mask = (const unsigned char*)d_in[2];
    const float* n1   = (const float*)d_in[3];
    const float* wq   = (const float*)d_in[4];
    const float* wk   = (const float*)d_in[5];
    const float* wv   = (const float*)d_in[6];
    const float* wo   = (const float*)d_in[7];
    const float* n2   = (const float*)d_in[8];
    const float* w1   = (const float*)d_in[9];
    const float* b1   = (const float*)d_in[10];
    const float* w2   = (const float*)d_in[11];
    const float* b2   = (const float*)d_in[12];

    size_t off = 0;
    auto wsalloc = [&](size_t bytes) {
        void* p = (char*)d_ws + off;
        off += (bytes + 255) & ~(size_t)255;
        return p;
    };
    __bf16* wq_b = (__bf16*)wsalloc((size_t)CD * CD * 2);
    __bf16* wk_b = (__bf16*)wsalloc((size_t)CD * CD * 2);
    __bf16* wv_b = (__bf16*)wsalloc((size_t)CD * CD * 2);
    __bf16* wo_b = (__bf16*)wsalloc((size_t)CD * CD * 2);
    __bf16* w1_b = (__bf16*)wsalloc((size_t)CHID * CD * 2);
    __bf16* w2_b = (__bf16*)wsalloc((size_t)CD * CHID * 2);
    __bf16* x1   = (__bf16*)wsalloc((size_t)CM * CD * 2);
    __bf16* Qb   = (__bf16*)wsalloc((size_t)CM * CD * 2);
    __bf16* Kb   = (__bf16*)wsalloc((size_t)CM * CD * 2);
    __bf16* Vb   = (__bf16*)wsalloc((size_t)CM * CD * 2);
    __bf16* Yb   = (__bf16*)wsalloc((size_t)CM * CD * 2);
    float*  Ab   = (float*) wsalloc((size_t)CM * CD * 4);
    __bf16* Zb   = (__bf16*)wsalloc((size_t)CM * CD * 2);
    __bf16* Hb   = (__bf16*)wsalloc((size_t)CM * CHID * 2);

    // 1) weights -> bf16
    cast_bf16_kernel<<<2048, 256, 0, stream>>>(wq, wq_b, (long)CD * CD);
    cast_bf16_kernel<<<2048, 256, 0, stream>>>(wk, wk_b, (long)CD * CD);
    cast_bf16_kernel<<<2048, 256, 0, stream>>>(wv, wv_b, (long)CD * CD);
    cast_bf16_kernel<<<2048, 256, 0, stream>>>(wo, wo_b, (long)CD * CD);
    cast_bf16_kernel<<<4096, 256, 0, stream>>>(w1, w1_b, (long)CHID * CD);
    cast_bf16_kernel<<<4096, 256, 0, stream>>>(w2, w2_b, (long)CD * CHID);

    // 2) x1 = rmsnorm(H) * n1
    rmsnorm_kernel<<<CM, 256, 0, stream>>>(Hh, nullptr, n1, x1, CD);

    // 3) Q/K/V projections
    dim3 gQ(CM / 128, CD / 128);
    gemm_bf16_wmma<0><<<gQ, 256, 0, stream>>>(x1, wq_b, nullptr, nullptr, Qb, CM, CD, CD);
    gemm_bf16_wmma<0><<<gQ, 256, 0, stream>>>(x1, wk_b, nullptr, nullptr, Kb, CM, CD, CD);
    gemm_bf16_wmma<0><<<gQ, 256, 0, stream>>>(x1, wv_b, nullptr, nullptr, Vb, CM, CD, CD);

    // 4) flash attention -> Yb
    dim3 gF(CT / 64, CB * CNH);
    flash_attn_kernel<<<gF, 128, 0, stream>>>(Qb, Kb, Vb, mask, Yb);

    // 5) A = Y @ wo^T  (fp32)
    gemm_bf16_wmma<1><<<gQ, 256, 0, stream>>>(Yb, wo_b, nullptr, nullptr, Ab, CM, CD, CD);

    // 6) z = rmsnorm(H + A) * n2
    rmsnorm_kernel<<<CM, 256, 0, stream>>>(Hh, Ab, n2, Zb, CD);

    // 7) h = gelu(z @ w1^T + b1)
    dim3 g1(CM / 128, CHID / 128);
    gemm_bf16_wmma<2><<<g1, 256, 0, stream>>>(Zb, w1_b, b1, nullptr, Hb, CM, CHID, CD);

    // 8) out = A + (h @ w2^T + b2)
    dim3 g2(CM / 128, CD / 128);
    gemm_bf16_wmma<3><<<g2, 256, 0, stream>>>(Hb, w2_b, b2, Ab, d_out, CM, CD, CHID);
}
